// RGNNLayer_70523363000618
// MI455X (gfx1250) — matrix operand order repaired and checked by reading, not compiled, not run on hardware
//
#include <hip/hip_runtime.h>
#include <stdint.h>

#define Bdim 8
#define Tdim 16
#define Ndim 1024
#define Fdim 128
#define Kdim 4

typedef __attribute__((ext_vector_type(16))) __bf16 v16bf;
typedef __attribute__((ext_vector_type(8)))  float  v8f;
typedef __attribute__((ext_vector_type(4)))  float  f32x4;
typedef __attribute__((ext_vector_type(4)))  unsigned int u32x4;
typedef __attribute__((ext_vector_type(2)))  unsigned int u32x2;

union Frag {
  v16bf v;
  u32x4 q[2];
  unsigned int u[8];
};

__device__ __forceinline__ unsigned int pk_bf16(float a, float b) {
  unsigned int ua = __builtin_bit_cast(unsigned int, a);
  unsigned int ub = __builtin_bit_cast(unsigned int, b);
  ua = (ua + 0x7FFFu + ((ua >> 16) & 1u)) >> 16;  // round-to-nearest-even
  ub = (ub + 0x7FFFu + ((ub >> 16) & 1u)) >> 16;
  return (ua & 0xFFFFu) | (ub << 16);
}
__device__ __forceinline__ unsigned short f32_bf16(float a) {
  unsigned int ua = __builtin_bit_cast(unsigned int, a);
  return (unsigned short)((ua + 0x7FFFu + ((ua >> 16) & 1u)) >> 16);
}

__device__ __forceinline__ v8f wmma_bf16(v16bf a, v16bf b, v8f c) {
  // D = A(16x32 bf16) * B(32x16 bf16) + C(f32)
  return __builtin_amdgcn_wmma_f32_16x16x32_bf16(false, a, false, b, (short)0, c,
                                                 false, false);
}

// B-operand fragment: 32 contiguous bytes per lane from a pre-packed blob.
__device__ __forceinline__ v16bf load_b_frag(const unsigned short* p) {
  Frag f;
  f.q[0] = *(const u32x4*)p;
  f.q[1] = *(const u32x4*)(p + 8);
  return f.v;
}
// A-operand fragment from a row-major bf16 matrix: lane holds row (lane%16),
// K-cols [c0..c0+15] with c0 = kchunk*32 + (lane/16)*16.
__device__ __forceinline__ v16bf load_a_frag_bf(const unsigned short* p) {
  Frag f;
  f.q[0] = *(const u32x4*)p;        // cols c0..c0+7
  f.q[1] = *(const u32x4*)(p + 8);  // cols c0+8..c0+15
  return f.v;
}
// Same, but source is f32 row-major: convert in-register.
__device__ __forceinline__ v16bf load_a_frag_f32(const float* p) {
  Frag f;
#pragma unroll
  for (int i = 0; i < 4; ++i) f.u[i] = pk_bf16(p[2 * i], p[2 * i + 1]);
#pragma unroll
  for (int i = 0; i < 4; ++i) f.u[4 + i] = pk_bf16(p[8 + 2 * i], p[9 + 2 * i]);
  return f.v;
}

__device__ __forceinline__ v8f zero8() {
  v8f z = {0.f, 0.f, 0.f, 0.f, 0.f, 0.f, 0.f, 0.f};
  return z;
}

// f32 row-major -> bf16 row-major
__global__ void cvt_bf16_kernel(const float* __restrict__ src,
                                unsigned short* __restrict__ dst, int total) {
  int i = blockIdx.x * blockDim.x + threadIdx.x;
  if (i < total) dst[i] = f32_bf16(src[i]);
}

// Pack an f32 (R x C) matrix (batched) into the WMMA B-operand layout:
// tiles of 32(K) x 16(N); tile_id = nt*(R/32) + kt; within a tile,
// lane = (kk/16)*16 + n holds 16 contiguous bf16 with j = kk%16.
__global__ void pack_b_kernel(const float* __restrict__ src,
                              unsigned short* __restrict__ dst,
                              int R, int C, int total) {
  int stride = gridDim.x * blockDim.x;
  for (int i = blockIdx.x * blockDim.x + threadIdx.x; i < total; i += stride) {
    int mat = i / (R * C);
    int e = i - mat * (R * C);
    int tile = e >> 9;
    int rem = e & 511;
    int lanep = rem >> 4;
    int j = rem & 15;
    int ktiles = R >> 5;
    int nt = tile / ktiles;
    int kt = tile - nt * ktiles;
    int row = kt * 32 + ((lanep >> 4) << 4) + j;
    int col = nt * 16 + (lanep & 15);
    dst[i] = f32_bf16(src[(size_t)mat * R * C + (size_t)row * C + col]);
  }
}

// C_row(bf16) = A_row(bf16, 1024x1024) @ B_pack(1024x1024). One wave = 16x64 strip.
__global__ void __launch_bounds__(256)
pow_kernel(const unsigned short* __restrict__ A,
           const unsigned short* __restrict__ Bp,
           unsigned short* __restrict__ Cst) {
  int tid = threadIdx.x, lane = tid & 31, half = lane >> 4, lrow = lane & 15;
  int gw = blockIdx.x * 8 + (tid >> 5);          // 1024 waves
  int mt = gw >> 4, ns = gw & 15;

  v8f acc[4];
#pragma unroll
  for (int i = 0; i < 4; ++i) acc[i] = zero8();

  const unsigned short* arow = A + (size_t)(mt * 16 + lrow) * Ndim;
#pragma unroll 1
  for (int mc = 0; mc < 32; ++mc) {
    v16bf a = load_a_frag_bf(arow + mc * 32 + half * 16);
    __builtin_prefetch(arow + (mc + 1) * 32 + half * 8, 0, 1);
    v16bf bv[4];
#pragma unroll
    for (int n4 = 0; n4 < 4; ++n4)
      bv[n4] = load_b_frag(Bp + ((size_t)((ns * 4 + n4) * 32 + mc) << 9) + lane * 16);
#pragma unroll
    for (int n4 = 0; n4 < 4; ++n4)
      acc[n4] = wmma_bf16(a, bv[n4], acc[n4]);
  }
#pragma unroll
  for (int n4 = 0; n4 < 4; ++n4) {
    int col = (ns * 4 + n4) * 16 + lrow;
#pragma unroll
    for (int r = 0; r < 8; ++r) {
      int row = mt * 16 + half * 8 + r;
      Cst[(size_t)row * Ndim + col] = f32_bf16(acc[n4][r]);
    }
  }
}

// s[k,b] = x_t[b] @ W[k] + h[b] @ H[k]; wave = one 16x128 strip of N x F.
// Output written directly in packed B layout for the zpart-kernel.
__global__ void __launch_bounds__(256)
s_kernel(const float* __restrict__ x,
         const unsigned short* __restrict__ hbf,
         const unsigned short* __restrict__ Wp,
         const unsigned short* __restrict__ Hp,
         unsigned short* __restrict__ spack, int t) {
  int tid = threadIdx.x, lane = tid & 31, half = lane >> 4, lrow = lane & 15;
  int gw = blockIdx.x * 8 + (tid >> 5);          // 2048 waves
  int k = gw >> 9;
  int rem = gw & 511;
  int b = rem >> 6;
  int ntile = rem & 63;

  v8f acc[8];
#pragma unroll
  for (int i = 0; i < 8; ++i) acc[i] = zero8();

  int row = ntile * 16 + lrow;
  const float* xrow = x + (((size_t)b * Tdim + t) * Ndim + row) * Fdim;
  const unsigned short* hrow = hbf + ((size_t)b * Ndim + row) * Fdim;
  const unsigned short* wb = Wp + k * Fdim * Fdim;
  const unsigned short* hb = Hp + k * Fdim * Fdim;

#pragma unroll 1
  for (int fc = 0; fc < 4; ++fc) {
    int c0 = fc * 32 + half * 16;
    {
      v16bf ax = load_a_frag_f32(xrow + c0);
      v16bf bv[8];
#pragma unroll
      for (int g = 0; g < 8; ++g)
        bv[g] = load_b_frag(wb + ((g * 4 + fc) << 9) + lane * 16);
#pragma unroll
      for (int g = 0; g < 8; ++g)
        acc[g] = wmma_bf16(ax, bv[g], acc[g]);
    }
    {
      v16bf ah = load_a_frag_bf(hrow + c0);
      v16bf bv[8];
#pragma unroll
      for (int g = 0; g < 8; ++g)
        bv[g] = load_b_frag(hb + ((g * 4 + fc) << 9) + lane * 16);
#pragma unroll
      for (int g = 0; g < 8; ++g)
        acc[g] = wmma_bf16(ah, bv[g], acc[g]);
    }
  }

  // Store C tiles straight into B-pack layout: 16 contiguous bytes per lane.
  size_t sbase = (size_t)(k * 8 + b) * ((size_t)Ndim * Fdim);
  int kt = ntile >> 1;
  int lanep = ((ntile & 1) << 4) + lrow;
#pragma unroll
  for (int g = 0; g < 8; ++g) {
    u32x4 wq;
    wq.x = pk_bf16(acc[g][0], acc[g][1]);
    wq.y = pk_bf16(acc[g][2], acc[g][3]);
    wq.z = pk_bf16(acc[g][4], acc[g][5]);
    wq.w = pk_bf16(acc[g][6], acc[g][7]);
    unsigned short* dst =
        spack + sbase + ((size_t)(g * 32 + kt) << 9) + lanep * 16 + half * 8;
    *(u32x4*)dst = wq;
  }
}

// Cooperatively stage the 8 B tiles (8 KB) for one mc step into LDS.
// tid group g = tid/32 handles tile g; each thread copies 32 contiguous bytes.
__device__ __forceinline__ void stage_tiles(const unsigned short* __restrict__ sb,
                                            int mc, unsigned short (*dst)[512],
                                            int tid) {
  int g = tid >> 5;
  int l = tid & 31;
  const unsigned short* src = sb + ((size_t)(g * 32 + mc) << 9) + l * 16;
  u32x4 q0 = *(const u32x4*)src;
  u32x4 q1 = *(const u32x4*)(src + 8);
  *(u32x4*)&dst[g][l * 16] = q0;
  *(u32x4*)&dst[g][l * 16 + 8] = q1;
}

// Split-K partial: zpart[k][b] = L^{k+1} @ s[k,b]  (f32, no relu).
// Grid = b(8) x msuper(4) x k(4) = 128 blocks of 8 waves.
// Wave = 4 Mtiles x 4 Ntiles (16 accum tiles, 64x64 f32 in regs).
// B tiles double-buffered through LDS; staging of step i+1 overlaps the
// 16 WMMAs of step i.
__global__ void __launch_bounds__(256)
zpart_kernel(const unsigned short* __restrict__ Pp,
             const unsigned short* __restrict__ spack,
             float* __restrict__ zpart) {
  __shared__ unsigned short smem[2][8][512];   // 2 x 8KB
  int tid = threadIdx.x, lane = tid & 31, half = lane >> 4, lrow = lane & 15;
  int wv = tid >> 5;
  int msub = wv & 3, nhalf = wv >> 2;
  int k = blockIdx.x & 3;
  int msuper = (blockIdx.x >> 2) & 3;
  int b = blockIdx.x >> 4;
  int mtbase = msuper * 16 + msub * 4;
  const size_t NF = (size_t)Ndim * Fdim;

  v8f acc[16];
#pragma unroll
  for (int i = 0; i < 16; ++i) acc[i] = zero8();

  const unsigned short* sb = spack + (size_t)(k * 8 + b) * NF;
  const unsigned short* abase = Pp + (size_t)k * Ndim * Ndim;

  stage_tiles(sb, 0, smem[0], tid);
  __syncthreads();

  int buf = 0;
#pragma unroll 1
  for (int mc = 0; mc < 32; ++mc) {
    if (mc + 1 < 32) stage_tiles(sb, mc + 1, smem[buf ^ 1], tid);

    v16bf af[4];
#pragma unroll
    for (int m4 = 0; m4 < 4; ++m4)
      af[m4] = load_a_frag_bf(abase + (size_t)((mtbase + m4) * 16 + lrow) * Ndim +
                              mc * 32 + half * 16);

    v16bf bf4[4];
#pragma unroll
    for (int n4 = 0; n4 < 4; ++n4) {
      Frag f;
      const unsigned short* p = &smem[buf][nhalf * 4 + n4][lane * 16];
      f.q[0] = *(const u32x4*)p;
      f.q[1] = *(const u32x4*)(p + 8);
      bf4[n4] = f.v;
    }

#pragma unroll
    for (int m4 = 0; m4 < 4; ++m4)
#pragma unroll
      for (int n4 = 0; n4 < 4; ++n4)
        acc[m4 * 4 + n4] = wmma_bf16(af[m4], bf4[n4], acc[m4 * 4 + n4]);

    __syncthreads();
    buf ^= 1;
  }

  float* zbase = zpart + (size_t)(k * Bdim + b) * NF;
#pragma unroll
  for (int m4 = 0; m4 < 4; ++m4)
#pragma unroll
    for (int n4 = 0; n4 < 4; ++n4) {
      int col = (nhalf * 4 + n4) * 16 + lrow;
#pragma unroll
      for (int r = 0; r < 8; ++r) {
        int row = (mtbase + m4) * 16 + half * 8 + r;
        zbase[(size_t)row * Fdim + col] = acc[m4 * 4 + n4][r];
      }
    }
}

// h = relu(sum_k zpart[k]); write f32 to out[:, t] and bf16 to hbf.
__global__ void __launch_bounds__(256)
reduce_kernel(const float* __restrict__ zpart,
              float* __restrict__ out,
              unsigned short* __restrict__ hbf, int t) {
  const size_t NF = (size_t)Ndim * Fdim;
  int i = blockIdx.x * blockDim.x + threadIdx.x;   // over B*NF/4
  int perb = (int)(NF >> 2);
  int b = i / perb;
  size_t r = (size_t)(i - b * perb) * 4;

  f32x4 s = *(const f32x4*)(zpart + (size_t)b * NF + r);
#pragma unroll
  for (int k = 1; k < Kdim; ++k) {
    f32x4 v = *(const f32x4*)(zpart + (size_t)(k * Bdim + b) * NF + r);
    s.x += v.x; s.y += v.y; s.z += v.z; s.w += v.w;
  }
  s.x = s.x > 0.0f ? s.x : 0.0f;
  s.y = s.y > 0.0f ? s.y : 0.0f;
  s.z = s.z > 0.0f ? s.z : 0.0f;
  s.w = s.w > 0.0f ? s.w : 0.0f;

  *(f32x4*)(out + ((size_t)b * Tdim + t) * NF + r) = s;
  u32x2 h;
  h.x = pk_bf16(s.x, s.y);
  h.y = pk_bf16(s.z, s.w);
  *(u32x2*)(hbf + (size_t)b * NF + r) = h;
}

extern "C" void kernel_launch(void* const* d_in, const int* in_sizes, int n_in,
                              void* d_out, int out_size, void* d_ws, size_t ws_size,
                              hipStream_t stream) {
  (void)in_sizes; (void)n_in; (void)out_size;
  const float* x = (const float*)d_in[0];
  const float* L = (const float*)d_in[1];
  const float* W = (const float*)d_in[2];
  const float* H = (const float*)d_in[3];
  float* out = (float*)d_out;

  const size_t NN = (size_t)Ndim * Ndim;   // 1M
  const size_t FF = (size_t)Fdim * Fdim;   // 16K
  const size_t NF = (size_t)Ndim * Fdim;   // 128K

  unsigned short* Lpack = (unsigned short*)d_ws;                // NN
  unsigned short* Ppow  = Lpack + NN;                           // 4*NN (L^1..L^4, bf16 row-major)
  unsigned short* Wp    = Ppow + 4 * NN;                        // 4*FF packed
  unsigned short* Hp    = Wp + 4 * FF;                          // 4*FF packed
  unsigned short* spack = Hp + 4 * FF;                          // K*B*NF packed
  unsigned short* hbf   = spack + (size_t)Kdim * Bdim * NF;     // B*NF bf16 row-major
  float*          zpart = (float*)(hbf + (size_t)Bdim * NF);    // K*B*NF f32 partials

  size_t need = (5 * NN + 8 * FF + (size_t)(Kdim * Bdim + Bdim) * NF) * sizeof(unsigned short)
              + (size_t)Kdim * Bdim * NF * sizeof(float);
  if (ws_size < need) return;

  // One-time prep
  cvt_bf16_kernel<<<(unsigned)((NN + 255) / 256), 256, 0, stream>>>(L, Ppow, (int)NN);
  pack_b_kernel<<<1024, 256, 0, stream>>>(L, Lpack, Ndim, Ndim, (int)NN);
  pack_b_kernel<<<64, 256, 0, stream>>>(W, Wp, Fdim, Fdim, (int)(4 * FF));
  pack_b_kernel<<<64, 256, 0, stream>>>(H, Hp, Fdim, Fdim, (int)(4 * FF));
  hipMemsetAsync(hbf, 0, Bdim * NF * sizeof(unsigned short), stream);

  // Power chain: L^2, L^3, L^4
  for (int p = 1; p < Kdim; ++p)
    pow_kernel<<<128, 256, 0, stream>>>(Ppow + (size_t)(p - 1) * NN, Lpack,
                                        Ppow + (size_t)p * NN);

  // Recurrent time loop (stream ordering = sequential dependence)
  const unsigned reduce_blocks = (unsigned)((Bdim * NF / 4 + 255) / 256);
  for (int t = 0; t < Tdim; ++t) {
    s_kernel<<<256, 256, 0, stream>>>(x, hbf, Wp, Hp, spack, t);
    zpart_kernel<<<128, 256, 0, stream>>>(Ppow, spack, zpart);
    reduce_kernel<<<reduce_blocks, 256, 0, stream>>>(zpart, out, hbf, t);
  }
}
